// DAM_5179730559354
// MI455X (gfx1250) — matrix operand order repaired and checked by reading, not compiled, not run on hardware
//
#include <hip/hip_runtime.h>
#include <stdint.h>

// ---------------- CDNA5 WMMA types ----------------
typedef __attribute__((ext_vector_type(16))) __bf16 v16bf;
typedef __attribute__((ext_vector_type(8)))  float  v8f;

__device__ __forceinline__ uint16_t f2bf(float f) {
  uint32_t u = __builtin_bit_cast(uint32_t, f);
  u += 0x7FFFu + ((u >> 16) & 1u);           // round to nearest even
  return (uint16_t)(u >> 16);
}
__device__ __forceinline__ float bf2f(uint16_t h) {
  uint32_t u = ((uint32_t)h) << 16;
  return __builtin_bit_cast(float, u);
}

struct Bits256 { uint4 lo, hi; };

// A fragment: A stored row-major [M][K] (K contiguous), 16x32 tile at (row0, k).
// ISA 7.12.2: lanes 0-15 -> M=lane, K = k+0..7 / k+16..23; lanes 16-31 -> K +8.
__device__ __forceinline__ v16bf load_frag_a(const uint16_t* A, int lda,
                                             int row0, int k, int l, int hi) {
  const uint16_t* p = A + (size_t)(row0 + l) * lda + k + hi * 8;
  Bits256 r;
  r.lo = *(const uint4*)(p);        // e0..7
  r.hi = *(const uint4*)(p + 16);   // e8..15
  return __builtin_bit_cast(v16bf, r);
}

// B fragment: B stored transposed [Ncol][K] (K contiguous), 32x16 tile at (k, col0).
__device__ __forceinline__ v16bf load_frag_b(const uint16_t* Bt, int ldb,
                                             int col0, int k, int l, int hi) {
  const uint16_t* p = Bt + (size_t)(col0 + l) * ldb + k + hi * 8;
  Bits256 r;
  r.lo = *(const uint4*)(p);
  r.hi = *(const uint4*)(p + 16);
  return __builtin_bit_cast(v16bf, r);
}

// Register-blocked wave GEMM: one wave computes a (16*MT)x(16*NT) block.
// Per k-step: MT+NT fragment loads feed MT*NT wmmas (4x4 -> 1:2 load:wmma).
template <int MT, int NT>
__device__ __forceinline__ void gemm_block(const uint16_t* __restrict__ A, int lda,
                                           const uint16_t* __restrict__ Bt, int ldb,
                                           int row0, int col0, int K,
                                           v8f (&acc)[MT][NT]) {
  int lane = threadIdx.x & 31;
  int l = lane & 15, hi = lane >> 4;
  for (int k = 0; k < K; k += 32) {
    v16bf af[MT], bfr[NT];
#pragma unroll
    for (int i = 0; i < MT; i++) af[i] = load_frag_a(A, lda, row0 + 16 * i, k, l, hi);
#pragma unroll
    for (int j = 0; j < NT; j++) bfr[j] = load_frag_b(Bt, ldb, col0 + 16 * j, k, l, hi);
#pragma unroll
    for (int i = 0; i < MT; i++)
#pragma unroll
      for (int j = 0; j < NT; j++)
        acc[i][j] = __builtin_amdgcn_wmma_f32_16x16x32_bf16(
            false, af[i], false, bfr[j], (short)0, acc[i][j], false, false);
  }
}

__device__ __forceinline__ void store_tile_f32(float* C, int ldc, int row0, int col0,
                                               v8f a, int l, int hi) {
  float* p = C + (size_t)(row0 + hi * 8) * ldc + col0 + l;
#pragma unroll
  for (int i = 0; i < 8; i++) p[(size_t)i * ldc] = a[i];
}
__device__ __forceinline__ void store_tile_bf16(uint16_t* C, int ldc, int row0, int col0,
                                                v8f a, int l, int hi) {
  uint16_t* p = C + (size_t)(row0 + hi * 8) * ldc + col0 + l;
#pragma unroll
  for (int i = 0; i < 8; i++) p[(size_t)i * ldc] = f2bf(a[i]);
}

// ---------------- problem constants ----------------
#define BB 4
#define CC 512
#define C8 64
#define NN 4096   // H*W

// ---------------- elementwise / transpose ----------------
__global__ void k_cvt_bf16(const float* __restrict__ s, uint16_t* __restrict__ d, int n) {
  int i = blockIdx.x * blockDim.x + threadIdx.x;
  if (i < n) d[i] = f2bf(s[i]);
}

// x[b][c][n] f32 -> xh bf16 [b][c][n] and xT bf16 [b][n][c] (LDS tiled transpose)
__global__ void k_transpose(const float* __restrict__ x, uint16_t* __restrict__ xh,
                            uint16_t* __restrict__ xT) {
  __shared__ uint16_t tile[32][33];
  int b = blockIdx.z;
  int n0 = blockIdx.x * 32, c0 = blockIdx.y * 32;
  const float* xb = x + (size_t)b * CC * NN;
  uint16_t* xhb = xh + (size_t)b * CC * NN;
  uint16_t* xTb = xT + (size_t)b * NN * CC;
  int tx = threadIdx.x;
  for (int i = threadIdx.y; i < 32; i += 8) {
    uint16_t h = f2bf(xb[(size_t)(c0 + i) * NN + n0 + tx]);
    xhb[(size_t)(c0 + i) * NN + n0 + tx] = h;
    tile[i][tx] = h;
  }
  __syncthreads();
  for (int i = threadIdx.y; i < 32; i += 8)
    xTb[(size_t)(n0 + i) * CC + c0 + tx] = tile[tx][i];
}

// ---------------- GEMM kernels (128 threads = 4 waves, 64x64 block/wave) ------
// fbT/fcT[b][n][j] = sum_c x[c][n]*w[j][c] + bias[j]   (M=4096, N=64, K=512)
__global__ __launch_bounds__(128)
void k_gemm_proj(const uint16_t* __restrict__ xT, const uint16_t* __restrict__ wh,
                 const float* __restrict__ bias, uint16_t* __restrict__ outT) {
  int b = blockIdx.z, wave = threadIdx.x >> 5;
  int row0 = (blockIdx.x * 4 + wave) * 64, col0 = 0;
  v8f acc[4][4] = {};
  gemm_block<4, 4>(xT + (size_t)b * NN * CC, CC, wh, CC, row0, col0, CC, acc);
  int lane = threadIdx.x & 31, l = lane & 15, hi = lane >> 4;
  uint16_t* ob = outT + (size_t)b * NN * C8;
#pragma unroll
  for (int i = 0; i < 4; i++)
#pragma unroll
    for (int j = 0; j < 4; j++) {
      float bv = bias[col0 + 16 * j + l];
      v8f a = acc[i][j];
#pragma unroll
      for (int r = 0; r < 8; r++) a[r] += bv;
      store_tile_bf16(ob, C8, row0 + 16 * i, col0 + 16 * j, a, l, hi);
    }
}

// fd[b][c][m] = sum_c' wd[c][c']*x[c'][m] + bd[c]   (M=512, N=4096, K=512)
__global__ __launch_bounds__(128)
void k_gemm_fd(const uint16_t* __restrict__ wdh, const uint16_t* __restrict__ xT,
               const float* __restrict__ bd, uint16_t* __restrict__ fdh) {
  int b = blockIdx.z, wave = threadIdx.x >> 5;
  int row0 = (blockIdx.x * 2 + (wave >> 1)) * 64;
  int col0 = (blockIdx.y * 2 + (wave & 1)) * 64;
  v8f acc[4][4] = {};
  gemm_block<4, 4>(wdh, CC, xT + (size_t)b * NN * CC, CC, row0, col0, CC, acc);
  int lane = threadIdx.x & 31, l = lane & 15, hi = lane >> 4;
  uint16_t* ob = fdh + (size_t)b * CC * NN;
#pragma unroll
  for (int i = 0; i < 4; i++)
#pragma unroll
    for (int j = 0; j < 4; j++) {
      v8f a = acc[i][j];
#pragma unroll
      for (int r = 0; r < 8; r++) a[r] += bd[row0 + 16 * i + hi * 8 + r];
      store_tile_bf16(ob, NN, row0 + 16 * i, col0 + 16 * j, a, l, hi);
    }
}

// att[b][c][d] = sum_n x[c][n]*x[d][n]   (M=N=512, K=4096)
__global__ __launch_bounds__(128)
void k_gemm_att(const uint16_t* __restrict__ xh, float* __restrict__ att) {
  int b = blockIdx.z, wave = threadIdx.x >> 5;
  int row0 = (blockIdx.x * 2 + (wave >> 1)) * 64;
  int col0 = (blockIdx.y * 2 + (wave & 1)) * 64;
  const uint16_t* X = xh + (size_t)b * CC * NN;
  v8f acc[4][4] = {};
  gemm_block<4, 4>(X, NN, X, NN, row0, col0, NN, acc);
  int lane = threadIdx.x & 31, l = lane & 15, hi = lane >> 4;
  float* ob = att + (size_t)b * CC * CC;
#pragma unroll
  for (int i = 0; i < 4; i++)
#pragma unroll
    for (int j = 0; j < 4; j++)
      store_tile_f32(ob, CC, row0 + 16 * i, col0 + 16 * j, acc[i][j], l, hi);
}

// softmax_d(rowmax - a) == exp(rowmin - a)/sum  (rowmax cancels)
__global__ void k_chan_softmax(const float* __restrict__ att, uint16_t* __restrict__ attC) {
  int row = blockIdx.x;                       // B*C rows
  const float* a = att + (size_t)row * CC;
  uint16_t* o = attC + (size_t)row * CC;
  int t = threadIdx.x;                        // 256 threads, 2 cols each
  float v0 = a[t], v1 = a[t + 256];
  __shared__ float red[256];
  red[t] = fminf(v0, v1); __syncthreads();
  for (int s = 128; s > 0; s >>= 1) { if (t < s) red[t] = fminf(red[t], red[t + s]); __syncthreads(); }
  float mn = red[0]; __syncthreads();
  float e0 = __expf(mn - v0), e1 = __expf(mn - v1);
  red[t] = e0 + e1; __syncthreads();
  for (int s = 128; s > 0; s >>= 1) { if (t < s) red[t] += red[t + s]; __syncthreads(); }
  float inv = 1.f / red[0];
  o[t] = f2bf(e0 * inv); o[t + 256] = f2bf(e1 * inv);
}

// featE[b][c][n] = sum_d attC[c][d]*x[d][n]   (M=512, N=4096, K=512)
__global__ __launch_bounds__(128)
void k_gemm_feate(const uint16_t* __restrict__ attC, const uint16_t* __restrict__ xT,
                  float* __restrict__ featE) {
  int b = blockIdx.z, wave = threadIdx.x >> 5;
  int row0 = (blockIdx.x * 2 + (wave >> 1)) * 64;
  int col0 = (blockIdx.y * 2 + (wave & 1)) * 64;
  v8f acc[4][4] = {};
  gemm_block<4, 4>(attC + (size_t)b * CC * CC, CC, xT + (size_t)b * NN * CC, CC,
                   row0, col0, CC, acc);
  int lane = threadIdx.x & 31, l = lane & 15, hi = lane >> 4;
  float* ob = featE + (size_t)b * CC * NN;
#pragma unroll
  for (int i = 0; i < 4; i++)
#pragma unroll
    for (int j = 0; j < 4; j++)
      store_tile_f32(ob, NN, row0 + 16 * i, col0 + 16 * j, acc[i][j], l, hi);
}

// S[n][m] = sum_j fbT[n][j]*fcT[m][j]   (M=N=4096, K=64; per batch)
__global__ __launch_bounds__(128)
void k_gemm_logits(const uint16_t* __restrict__ fbT, const uint16_t* __restrict__ fcT,
                   uint16_t* __restrict__ SP) {
  int wave = threadIdx.x >> 5;
  int row0 = (blockIdx.x * 2 + (wave >> 1)) * 64;
  int col0 = (blockIdx.y * 2 + (wave & 1)) * 64;
  v8f acc[4][4] = {};
  gemm_block<4, 4>(fbT, C8, fcT, C8, row0, col0, C8, acc);
  int lane = threadIdx.x & 31, l = lane & 15, hi = lane >> 4;
#pragma unroll
  for (int i = 0; i < 4; i++)
#pragma unroll
    for (int j = 0; j < 4; j++)
      store_tile_bf16(SP, NN, row0 + 16 * i, col0 + 16 * j, acc[i][j], l, hi);
}

// in-place row softmax over 4096 bf16 values
__global__ void k_pos_softmax(uint16_t* __restrict__ SP) {
  uint16_t* r = SP + (size_t)blockIdx.x * NN;
  int t = threadIdx.x;                        // 256 threads, 16 vals each
  float vals[16]; float mx = -3.4e38f;
#pragma unroll
  for (int i = 0; i < 16; i++) { vals[i] = bf2f(r[t + i * 256]); mx = fmaxf(mx, vals[i]); }
  __shared__ float red[256];
  red[t] = mx; __syncthreads();
  for (int s = 128; s > 0; s >>= 1) { if (t < s) red[t] = fmaxf(red[t], red[t + s]); __syncthreads(); }
  mx = red[0]; __syncthreads();
  float sum = 0.f;
#pragma unroll
  for (int i = 0; i < 16; i++) { vals[i] = __expf(vals[i] - mx); sum += vals[i]; }
  red[t] = sum; __syncthreads();
  for (int s = 128; s > 0; s >>= 1) { if (t < s) red[t] += red[t + s]; __syncthreads(); }
  float inv = 1.f / red[0];
#pragma unroll
  for (int i = 0; i < 16; i++) r[t + i * 256] = f2bf(vals[i] * inv);
}

// out[c][n] = beta*featE + alpha*(sum_m fd[c][m]*P[n][m]) + 2*x   (M=512,N=4096,K=4096)
__global__ __launch_bounds__(128)
void k_gemm_featp_final(const uint16_t* __restrict__ fdh, const uint16_t* __restrict__ SP,
                        const float* __restrict__ featE, const float* __restrict__ x,
                        const float* __restrict__ alpha, const float* __restrict__ beta,
                        float* __restrict__ out) {
  int wave = threadIdx.x >> 5;
  int row0 = (blockIdx.x * 2 + (wave >> 1)) * 64;
  int col0 = (blockIdx.y * 2 + (wave & 1)) * 64;
  v8f acc[4][4] = {};
  gemm_block<4, 4>(fdh, NN, SP, NN, row0, col0, NN, acc);
  int lane = threadIdx.x & 31, l = lane & 15, hi = lane >> 4;
  float al = alpha[0], be = beta[0];
#pragma unroll
  for (int i = 0; i < 4; i++)
#pragma unroll
    for (int j = 0; j < 4; j++) {
      size_t base = (size_t)(row0 + 16 * i + hi * 8) * NN + col0 + 16 * j + l;
#pragma unroll
      for (int r = 0; r < 8; r++) {
        size_t idx = base + (size_t)r * NN;
        out[idx] = be * featE[idx] + al * acc[i][j][r] + 2.f * x[idx];
      }
    }
}

// ---------------- host ----------------
extern "C" void kernel_launch(void* const* d_in, const int* in_sizes, int n_in,
                              void* d_out, int out_size, void* d_ws, size_t ws_size,
                              hipStream_t stream) {
  (void)in_sizes; (void)n_in; (void)out_size; (void)ws_size;
  const float* x     = (const float*)d_in[0];
  const float* wb    = (const float*)d_in[1];
  const float* bb    = (const float*)d_in[2];
  const float* wc    = (const float*)d_in[3];
  const float* bc    = (const float*)d_in[4];
  const float* wd    = (const float*)d_in[5];
  const float* bd    = (const float*)d_in[6];
  const float* alpha = (const float*)d_in[7];
  const float* beta  = (const float*)d_in[8];
  float* out = (float*)d_out;

  char* ws = (char*)d_ws; size_t off = 0;
  auto alloc = [&](size_t bytes) -> char* {
    char* p = ws + off; off = (off + bytes + 255) & ~(size_t)255; return p;
  };
  uint16_t* xh   = (uint16_t*)alloc((size_t)BB * CC * NN * 2);  // 16 MB
  uint16_t* xT   = (uint16_t*)alloc((size_t)BB * NN * CC * 2);  // 16 MB
  uint16_t* wbh  = (uint16_t*)alloc((size_t)C8 * CC * 2);
  uint16_t* wch  = (uint16_t*)alloc((size_t)C8 * CC * 2);
  uint16_t* wdh  = (uint16_t*)alloc((size_t)CC * CC * 2);
  uint16_t* fbT  = (uint16_t*)alloc((size_t)BB * NN * C8 * 2);  // 2 MB
  uint16_t* fcT  = (uint16_t*)alloc((size_t)BB * NN * C8 * 2);  // 2 MB
  uint16_t* fdh  = (uint16_t*)alloc((size_t)BB * CC * NN * 2);  // 16 MB
  float*    attF = (float*)   alloc((size_t)BB * CC * CC * 4);  // 4 MB
  uint16_t* attC = (uint16_t*)alloc((size_t)BB * CC * CC * 2);  // 2 MB
  float*    fE   = (float*)   alloc((size_t)BB * CC * NN * 4);  // 32 MB
  uint16_t* SP   = (uint16_t*)alloc((size_t)NN * NN * 2);       // 32 MB (per-batch reuse)

  // bf16 conversions
  k_cvt_bf16<<<(C8 * CC + 255) / 256, 256, 0, stream>>>(wb, wbh, C8 * CC);
  k_cvt_bf16<<<(C8 * CC + 255) / 256, 256, 0, stream>>>(wc, wch, C8 * CC);
  k_cvt_bf16<<<(CC * CC + 255) / 256, 256, 0, stream>>>(wd, wdh, CC * CC);
  k_transpose<<<dim3(NN / 32, CC / 32, BB), dim3(32, 8), 0, stream>>>(x, xh, xT);

  // projections + channel branch
  k_gemm_proj<<<dim3(NN / 256, 1, BB), 128, 0, stream>>>(xT, wbh, bb, fbT);
  k_gemm_proj<<<dim3(NN / 256, 1, BB), 128, 0, stream>>>(xT, wch, bc, fcT);
  k_gemm_fd  <<<dim3(CC / 128, NN / 128, BB), 128, 0, stream>>>(wdh, xT, bd, fdh);
  k_gemm_att <<<dim3(CC / 128, CC / 128, BB), 128, 0, stream>>>(xh, attF);
  k_chan_softmax<<<dim3(BB * CC), 256, 0, stream>>>(attF, attC);
  k_gemm_feate<<<dim3(CC / 128, NN / 128, BB), 128, 0, stream>>>(attC, xT, fE);

  // position branch, per batch (reuses SP to cap workspace at ~123 MB)
  for (int b = 0; b < BB; b++) {
    k_gemm_logits<<<dim3(NN / 128, NN / 128), 128, 0, stream>>>(
        fbT + (size_t)b * NN * C8, fcT + (size_t)b * NN * C8, SP);
    k_pos_softmax<<<dim3(NN), 256, 0, stream>>>(SP);
    k_gemm_featp_final<<<dim3(CC / 128, NN / 128), 128, 0, stream>>>(
        fdh + (size_t)b * CC * NN, SP, fE + (size_t)b * CC * NN,
        x + (size_t)b * CC * NN, alpha, beta, out + (size_t)b * CC * NN);
  }
}
